// CPUMoE_22995254902972
// MI455X (gfx1250) — matrix operand order repaired and checked by reading, not compiled, not run on hardware
//
#include <hip/hip_runtime.h>
#include <hip/hip_bf16.h>

// ---------------------------------------------------------------------------
// MoE (DeepSeek-style, E=8, top-2, swiglu) for gfx1250 using bf16 WMMA.
// B=4, S=2048, H=2048, E=8, K=2, F=1408.  T = B*S = 8192 tokens.
// ---------------------------------------------------------------------------

typedef __attribute__((ext_vector_type(16))) __bf16 v16bf;
typedef __attribute__((ext_vector_type(8)))  float  v8f;

using u16 = unsigned short;
using u32 = unsigned int;

static constexpr int Tn = 8192;   // tokens
static constexpr int Hd = 2048;   // hidden
static constexpr int Ed = 8;      // experts
static constexpr int Kd = 2;      // top-k
static constexpr int Fd = 1408;   // moe inter dim

static constexpr int TM = 64;     // token tile
static constexpr int TN = 64;     // out-col tile
static constexpr int KK = 32;     // WMMA K step (bf16)
static constexpr int LSTR = 34;   // LDS row stride in u16 (padded vs 32)

__device__ __forceinline__ u16 f2bf(float f) {
    u32 u = __builtin_bit_cast(u32, f);
    u32 r = u + 0x7FFFu + ((u >> 16) & 1u);   // round-to-nearest-even
    return (u16)(r >> 16);
}

union FragBF { u32 u[8]; v16bf v; };

// ---------------------------------------------------------------------------
// fp32 -> bf16 bulk convert (grid-stride)
// ---------------------------------------------------------------------------
__global__ void cvt_f32_bf16_kernel(const float* __restrict__ src,
                                    u16* __restrict__ dst, int n) {
    int i = blockIdx.x * blockDim.x + threadIdx.x;
    int stride = gridDim.x * blockDim.x;
    for (; i < n; i += stride) dst[i] = f2bf(src[i]);
}

__global__ void zero_f32_kernel(float* __restrict__ dst, int n) {
    int i = blockIdx.x * blockDim.x + threadIdx.x;
    int stride = gridDim.x * blockDim.x;
    for (; i < n; i += stride) dst[i] = 0.0f;
}

__global__ void zero_i32_kernel(int* __restrict__ dst, int n) {
    int i = blockIdx.x * blockDim.x + threadIdx.x;
    if (i < n) dst[i] = 0;
}

// ---------------------------------------------------------------------------
// Router: one wave32 per token. logits = x @ router_w^T, softmax, top-2,
// normalized affinities. Counts accumulated with integer atomics
// (value-deterministic regardless of order).
// ---------------------------------------------------------------------------
__global__ void router_kernel(const float* __restrict__ x,
                              const float* __restrict__ rw,
                              int*   __restrict__ topk_idx,
                              float* __restrict__ topk_w,
                              int*   __restrict__ counts) {
    int gtid = blockIdx.x * blockDim.x + threadIdx.x;
    int t    = gtid >> 5;            // token = global wave id (wave32)
    int lane = threadIdx.x & 31;
    if (t >= Tn) return;

    const float* xt = x + (size_t)t * Hd;
    float acc[Ed];
#pragma unroll
    for (int e = 0; e < Ed; ++e) acc[e] = 0.0f;

    for (int h = lane; h < Hd; h += 32) {
        float xv = xt[h];
#pragma unroll
        for (int e = 0; e < Ed; ++e) acc[e] += xv * rw[e * Hd + h];
    }
#pragma unroll
    for (int off = 16; off > 0; off >>= 1) {
#pragma unroll
        for (int e = 0; e < Ed; ++e) acc[e] += __shfl_xor(acc[e], off, 32);
    }
    if (lane == 0) {
        float mx = acc[0];
#pragma unroll
        for (int e = 1; e < Ed; ++e) mx = fmaxf(mx, acc[e]);
        float p[Ed], s = 0.0f;
#pragma unroll
        for (int e = 0; e < Ed; ++e) { p[e] = __expf(acc[e] - mx); s += p[e]; }
        float inv_s = 1.0f / s;
#pragma unroll
        for (int e = 0; e < Ed; ++e) p[e] *= inv_s;
        // top-2, ties -> lowest index (matches jax.lax.top_k)
        int i0 = 0;
#pragma unroll
        for (int e = 1; e < Ed; ++e) if (p[e] > p[i0]) i0 = e;
        int i1 = (i0 == 0) ? 1 : 0;
#pragma unroll
        for (int e = 0; e < Ed; ++e) if (e != i0 && p[e] > p[i1]) i1 = e;
        float w0 = p[i0], w1 = p[i1];
        float inv = 1.0f / (w0 + w1 + 1e-20f);
        topk_idx[2 * t]     = i0;
        topk_idx[2 * t + 1] = i1;
        topk_w[2 * t]       = w0 * inv;
        topk_w[2 * t + 1]   = w1 * inv;
        atomicAdd(&counts[i0], 1);
        atomicAdd(&counts[i1], 1);
    }
}

// Exclusive prefix over the 8 expert counts.
__global__ void offsets_kernel(const int* __restrict__ counts,
                               int* __restrict__ offsets) {
    if (threadIdx.x == 0 && blockIdx.x == 0) {
        int s = 0;
#pragma unroll
        for (int e = 0; e < Ed; ++e) { offsets[e] = s; s += counts[e]; }
    }
}

// ---------------------------------------------------------------------------
// Deterministic compaction: one 256-thread workgroup per expert walks tokens
// in order and appends matching tokens (LDS Hillis-Steele scan).
// ---------------------------------------------------------------------------
__global__ void scatter_kernel(const int* __restrict__ topk_idx,
                               const float* __restrict__ topk_w,
                               const int* __restrict__ offsets,
                               int*   __restrict__ list_token,
                               float* __restrict__ list_w) {
    const int e   = blockIdx.x;
    const int tid = threadIdx.x;
    const int seg = offsets[e];
    __shared__ int sc[256];
    __shared__ int sbase;
    if (tid == 0) sbase = 0;
    __syncthreads();

    for (int base = 0; base < Tn; base += 256) {
        int t = base + tid;
        int i0 = topk_idx[2 * t], i1 = topk_idx[2 * t + 1];
        int sel = 0; float w = 0.0f;
        if (i0 == e)      { sel = 1; w = topk_w[2 * t]; }
        else if (i1 == e) { sel = 1; w = topk_w[2 * t + 1]; }
        sc[tid] = sel;
        __syncthreads();
#pragma unroll
        for (int off = 1; off < 256; off <<= 1) {
            int v = (tid >= off) ? sc[tid - off] : 0;
            __syncthreads();
            sc[tid] += v;
            __syncthreads();
        }
        if (sel) {
            int pos = sbase + sc[tid] - 1;
            list_token[seg + pos] = t;
            list_w[seg + pos]     = w;
        }
        __syncthreads();
        if (tid == 255) sbase += sc[255];
        __syncthreads();
    }
}

// ---------------------------------------------------------------------------
// WMMA fragment builders (CDNA5 ISA §7.12.2, wave32, 16-bit):
//   A 16x32: lane m=lane&15, half=lane>>4; vgpr j: K = (j/4)*16 + half*8 + (j%4)*2
//   B 32x16: lane n=lane&15, half=lane>>4; vgpr j: K = (j/4)*16 + half*8 + (j%4)*2
//   C 16x16: lane n=lane&15, half=lane>>4; vgpr r: M = half*8 + r
// ---------------------------------------------------------------------------
__device__ __forceinline__ v16bf load_fragA(const u16* lds, int mrow0,
                                            int lane) {
    const int m = mrow0 + (lane & 15);
    const int half = lane >> 4;
    FragBF f;
#pragma unroll
    for (int j = 0; j < 8; ++j) {
        int k = ((j >> 2) << 4) + (half << 3) + ((j & 3) << 1);
        f.u[j] = *(const u32*)&lds[m * LSTR + k];
    }
    return f.v;
}

__device__ __forceinline__ v16bf load_fragB(const u16* lds, int nrow0,
                                            int lane) {
    const int n = nrow0 + (lane & 15);
    const int half = lane >> 4;
    FragBF f;
#pragma unroll
    for (int j = 0; j < 8; ++j) {
        int k = ((j >> 2) << 4) + (half << 3) + ((j & 3) << 1);
        f.u[j] = *(const u32*)&lds[n * LSTR + k];
    }
    return f.v;
}

__device__ __forceinline__ v8f wmma_bf16(v16bf a, v16bf b, v8f c) {
    return __builtin_amdgcn_wmma_f32_16x16x32_bf16(
        /*neg_a=*/false, a, /*neg_b=*/false, b,
        /*c_mod=*/(short)0, c, /*reuse_a=*/false, /*reuse_b=*/false);
}

// ---------------------------------------------------------------------------
// Pass 1: h = route_w * silu(x @ gate^T) * (x @ up^T)  -> bf16 h-buffer.
// grid: (F/64, ceil(T/64), E), block 256 (8 waves). Wave w owns output slab
// rows (w&3)*16..+15, cols ((w>>2)*2)*16..+31, for both G and U accumulators.
// ---------------------------------------------------------------------------
__global__ void __launch_bounds__(256)
pass1_gate_up_kernel(const u16* __restrict__ xb,
                     const u16* __restrict__ gate_b,
                     const u16* __restrict__ up_b,
                     const int* __restrict__ counts,
                     const int* __restrict__ offsets,
                     const int* __restrict__ list_token,
                     const float* __restrict__ list_w,
                     u16* __restrict__ hbuf) {
    const int e   = blockIdx.z;
    const int cnt = counts[e];
    const int m0  = blockIdx.y * TM;
    if (m0 >= cnt) return;
    const int f0  = blockIdx.x * TN;
    const int seg = offsets[e];

    __shared__ u16 Xs[TM * LSTR];
    __shared__ u16 Gs[TN * LSTR];
    __shared__ u16 Us[TN * LSTR];

    const int tid  = threadIdx.x;
    const int wave = tid >> 5;
    const int lane = tid & 31;
    const int mi   = wave & 3;           // 16-row slab within the 64-token tile
    const int njb  = (wave >> 2) << 1;   // first of two 16-col slabs
    const int n    = lane & 15;
    const int half = lane >> 4;

    // cooperative loader role: 64 rows x 4 chunks of 8 halves (16B)
    const int lrow = tid >> 2;
    const int lk   = (tid & 3) << 3;
    const int trow = m0 + lrow;
    const int tok  = list_token[seg + (trow < cnt ? trow : 0)];
    const u16* xsrc = xb    + (size_t)tok * Hd + lk;
    const u16* gsrc = gate_b + ((size_t)e * Fd + f0 + lrow) * Hd + lk;
    const u16* usrc = up_b   + ((size_t)e * Fd + f0 + lrow) * Hd + lk;

    v8f accG0 = {}, accG1 = {}, accU0 = {}, accU1 = {};

    for (int h0 = 0; h0 < Hd; h0 += KK) {
        uint4 xv = *(const uint4*)(xsrc + h0);
        uint4 gv = *(const uint4*)(gsrc + h0);
        uint4 uv = *(const uint4*)(usrc + h0);
        __syncthreads();   // previous iteration's fragments consumed
        u32* xd = (u32*)&Xs[lrow * LSTR + lk];
        xd[0] = xv.x; xd[1] = xv.y; xd[2] = xv.z; xd[3] = xv.w;
        u32* gd = (u32*)&Gs[lrow * LSTR + lk];
        gd[0] = gv.x; gd[1] = gv.y; gd[2] = gv.z; gd[3] = gv.w;
        u32* ud = (u32*)&Us[lrow * LSTR + lk];
        ud[0] = uv.x; ud[1] = uv.y; ud[2] = uv.z; ud[3] = uv.w;
        __syncthreads();

        v16bf a   = load_fragA(Xs, mi * 16, lane);
        v16bf bg0 = load_fragB(Gs, njb * 16, lane);
        v16bf bg1 = load_fragB(Gs, (njb + 1) * 16, lane);
        v16bf bu0 = load_fragB(Us, njb * 16, lane);
        v16bf bu1 = load_fragB(Us, (njb + 1) * 16, lane);
        accG0 = wmma_bf16(a, bg0, accG0);
        accG1 = wmma_bf16(a, bg1, accG1);
        accU0 = wmma_bf16(a, bu0, accU0);
        accU1 = wmma_bf16(a, bu1, accU1);
    }

    // epilogue: silu(g)*u, pre-scaled by routing weight, bf16 store
#pragma unroll
    for (int r = 0; r < 8; ++r) {
        const int m   = (half << 3) + r;
        const int row = m0 + mi * 16 + m;
        if (row < cnt) {
            const float w = list_w[seg + row];
            const size_t base = (size_t)(seg + row) * Fd + f0;
            float g0 = accG0[r];
            float h0v = w * (g0 / (1.0f + __expf(-g0))) * accU0[r];
            hbuf[base + njb * 16 + n] = f2bf(h0v);
            float g1 = accG1[r];
            float h1v = w * (g1 / (1.0f + __expf(-g1))) * accU1[r];
            hbuf[base + (njb + 1) * 16 + n] = f2bf(h1v);
        }
    }
}

// ---------------------------------------------------------------------------
// Pass 2: out[token] += h_row @ down^T  for one expert. Launched once per
// expert sequentially on the stream -> deterministic accumulation order.
// grid: (H/64, ceil(T/64)), block 256.
// ---------------------------------------------------------------------------
__global__ void __launch_bounds__(256)
pass2_down_kernel(const u16* __restrict__ hbuf,
                  const u16* __restrict__ down_b,
                  const int* __restrict__ counts,
                  const int* __restrict__ offsets,
                  const int* __restrict__ list_token,
                  float* __restrict__ out, int e) {
    const int cnt = counts[e];
    const int m0  = blockIdx.y * TM;
    if (m0 >= cnt) return;
    const int hc0 = blockIdx.x * TN;   // output hidden-column tile
    const int seg = offsets[e];

    __shared__ u16 Hs[TM * LSTR];
    __shared__ u16 Ds[TN * LSTR];

    const int tid  = threadIdx.x;
    const int wave = tid >> 5;
    const int lane = tid & 31;
    const int mi   = wave & 3;
    const int njb  = (wave >> 2) << 1;
    const int n    = lane & 15;
    const int half = lane >> 4;

    const int lrow = tid >> 2;
    const int lk   = (tid & 3) << 3;
    const int trow = m0 + lrow;
    const int hrow = seg + (trow < cnt ? trow : 0);
    const u16* hsrc = hbuf   + (size_t)hrow * Fd + lk;
    const u16* dsrc = down_b + ((size_t)e * Hd + hc0 + lrow) * Fd + lk;

    v8f acc0 = {}, acc1 = {};

    for (int f0 = 0; f0 < Fd; f0 += KK) {   // 44 K-steps
        uint4 hv = *(const uint4*)(hsrc + f0);
        uint4 dv = *(const uint4*)(dsrc + f0);
        __syncthreads();
        u32* hd = (u32*)&Hs[lrow * LSTR + lk];
        hd[0] = hv.x; hd[1] = hv.y; hd[2] = hv.z; hd[3] = hv.w;
        u32* dd = (u32*)&Ds[lrow * LSTR + lk];
        dd[0] = dv.x; dd[1] = dv.y; dd[2] = dv.z; dd[3] = dv.w;
        __syncthreads();

        v16bf a  = load_fragA(Hs, mi * 16, lane);
        v16bf b0 = load_fragB(Ds, njb * 16, lane);
        v16bf b1 = load_fragB(Ds, (njb + 1) * 16, lane);
        acc0 = wmma_bf16(a, b0, acc0);
        acc1 = wmma_bf16(a, b1, acc1);
    }

#pragma unroll
    for (int r = 0; r < 8; ++r) {
        const int m   = (half << 3) + r;
        const int row = m0 + mi * 16 + m;
        if (row < cnt) {
            const int tok = list_token[seg + row];
            float* o = out + (size_t)tok * Hd + hc0;
            o[njb * 16 + n]       += acc0[r];
            o[(njb + 1) * 16 + n] += acc1[r];
        }
    }
}

// ---------------------------------------------------------------------------
// Host-side launcher
// ---------------------------------------------------------------------------
extern "C" void kernel_launch(void* const* d_in, const int* in_sizes, int n_in,
                              void* d_out, int out_size, void* d_ws, size_t ws_size,
                              hipStream_t stream) {
    const float* x    = (const float*)d_in[0];   // [T, H]
    const float* rw   = (const float*)d_in[1];   // [E, H]
    const float* gw   = (const float*)d_in[2];   // [E, F, H]
    const float* uw   = (const float*)d_in[3];   // [E, F, H]
    const float* dw   = (const float*)d_in[4];   // [E, H, F]
    float* out        = (float*)d_out;           // [T, H]

    // workspace carve-up (256B aligned)
    char* p = (char*)d_ws;
    auto alloc = [&](size_t bytes) {
        char* r = p;
        p += (bytes + 255) & ~(size_t)255;
        return r;
    };
    u16*   xb        = (u16*)  alloc((size_t)Tn * Hd * 2);         // 33.5 MB
    u16*   gate_b    = (u16*)  alloc((size_t)Ed * Fd * Hd * 2);    // 46.1 MB
    u16*   up_b      = (u16*)  alloc((size_t)Ed * Fd * Hd * 2);    // 46.1 MB
    u16*   down_b    = (u16*)  alloc((size_t)Ed * Hd * Fd * 2);    // 46.1 MB
    int*   topk_idx  = (int*)  alloc((size_t)Tn * Kd * 4);
    float* topk_w    = (float*)alloc((size_t)Tn * Kd * 4);
    int*   counts    = (int*)  alloc(Ed * 4);
    int*   offsets   = (int*)  alloc(Ed * 4);
    int*   list_tok  = (int*)  alloc((size_t)Tn * Kd * 4);
    float* list_w    = (float*)alloc((size_t)Tn * Kd * 4);
    u16*   hbuf      = (u16*)  alloc((size_t)Tn * Kd * Fd * 2);    // 46.1 MB

    // 0) reset per-call metadata (ws is not re-poisoned but we must not rely
    //    on stale state); counts must start at zero for the atomics.
    zero_i32_kernel<<<1, 32, 0, stream>>>(counts, Ed);

    // 1) fp32 -> bf16 conversions
    {
        int nx = Tn * Hd;
        cvt_f32_bf16_kernel<<<2048, 256, 0, stream>>>(x, xb, nx);
        int nw = Ed * Fd * Hd;
        cvt_f32_bf16_kernel<<<4096, 256, 0, stream>>>(gw, gate_b, nw);
        cvt_f32_bf16_kernel<<<4096, 256, 0, stream>>>(uw, up_b, nw);
        cvt_f32_bf16_kernel<<<4096, 256, 0, stream>>>(dw, down_b, nw);
    }

    // 2) router (one wave32 per token: 8192 waves -> 1024 blocks of 256)
    router_kernel<<<(Tn * 32) / 256, 256, 0, stream>>>(x, rw, topk_idx, topk_w,
                                                       counts);

    // 3) segment offsets + deterministic per-expert compaction
    offsets_kernel<<<1, 32, 0, stream>>>(counts, offsets);
    scatter_kernel<<<Ed, 256, 0, stream>>>(topk_idx, topk_w, offsets,
                                           list_tok, list_w);

    // 4) pass 1: gate/up WMMA GEMMs + swiglu -> hbuf (bf16)
    {
        dim3 grid(Fd / TN, (Tn + TM - 1) / TM, Ed);
        pass1_gate_up_kernel<<<grid, 256, 0, stream>>>(
            xb, gate_b, up_b, counts, offsets, list_tok, list_w, hbuf);
    }

    // 5) zero output, then pass 2 per expert (stream-serialized => determinism)
    zero_f32_kernel<<<2048, 256, 0, stream>>>(out, Tn * Hd);
    for (int e = 0; e < Ed; ++e) {
        dim3 grid(Hd / TN, (Tn + TM - 1) / TM);
        pass2_down_kernel<<<grid, 256, 0, stream>>>(
            hbuf, down_b, counts, offsets, list_tok, out, e);
    }
}
